// NeuralTuringMachine_86019605004615
// MI455X (gfx1250) — matrix operand order repaired and checked by reading, not compiled, not run on hardware
//
#include <hip/hip_runtime.h>
#include <stdint.h>

// NTM forward for MI455X (gfx1250, wave32).
// Persistent kernel: 32 blocks x 256 threads; each block owns 16 batch rows
// for all 64 timesteps. LSTM + head GEMMs use v_wmma_f32_16x16x32_bf16 with
// pre-swizzled bf16 weight fragments streamed from L2 via global_load_b128.
// Memory matrix M ([512,256,32] f32 = 16MB) stays in L2 (192MB).

#define N_SLOTS 256
#define M_UNITS 32
#define CTRL    256
#define N_IN    8
#define N_OUT   8
#define ADDRW   38
#define T_STEPS 64
#define B_TILE  16
#define CLIPV   20.0f
#define EPSV    1e-8f

#define KC_TILES 10   // controller K: 296 padded to 320
#define NC_TILES 64   // controller N: 1024
#define KH_TILES 8    // heads K: 256
#define NH_TILES 9    // heads N: 140 padded to 144

typedef __attribute__((ext_vector_type(16))) __bf16 v16bf;
typedef __attribute__((ext_vector_type(8)))  float  v8f;

struct Frag { union { uint32_t u[8]; v16bf v; }; };

__device__ __forceinline__ float sigmoidf_(float x) { return 1.0f / (1.0f + expf(-x)); }
__device__ __forceinline__ float softplusf_(float x) { return (x > 20.0f) ? x : log1pf(expf(x)); }

__device__ __forceinline__ uint32_t pack_bf16(float a, float b) {
  uint32_t ua = __float_as_uint(a), ub = __float_as_uint(b);
  uint32_t ra = (ua + 0x7FFFu + ((ua >> 16) & 1u)) >> 16;
  uint32_t rb = (ub + 0x7FFFu + ((ub >> 16) & 1u)) >> 16;
  return (ra & 0xFFFFu) | (rb << 16);
}

// A-fragment (16x32 bf16, M=batch rows) gathered from a row-major bf16 LDS
// array with rsu32 u32-words per row. Layout (contiguous k order, same as the B packing):
// lane m(0-15)/half: VGPR g holds K pair at 2*g + 16*half.
__device__ __forceinline__ Frag load_a_lds(const uint32_t* base, int rsu32, int kt, int lane) {
  Frag f;
  int m = lane & 15, half = lane >> 4;
  const uint32_t* p = base + m * rsu32 + kt * 16 + 8 * half;
  uint4 lo = *(const uint4*)p;
  uint4 hi = *(const uint4*)(p + 4);
  f.u[0] = lo.x; f.u[1] = lo.y; f.u[2] = lo.z; f.u[3] = lo.w;
  f.u[4] = hi.x; f.u[5] = hi.y; f.u[6] = hi.z; f.u[7] = hi.w;
  return f;
}

// B-fragment (32x16 bf16) pre-swizzled in global: [tile][lane][8 u32].
__device__ __forceinline__ Frag load_b_glb(const uint32_t* tileBase, int lane) {
  Frag f;
  const uint4* p = (const uint4*)(tileBase + lane * 8);
  uint4 lo = p[0], hi = p[1];
  f.u[0] = lo.x; f.u[1] = lo.y; f.u[2] = lo.z; f.u[3] = lo.w;
  f.u[4] = hi.x; f.u[5] = hi.y; f.u[6] = hi.z; f.u[7] = hi.w;
  return f;
}

// ---------------------------------------------------------------------------
// Weight conversion: f32 -> bf16 B-fragments in contiguous k order (k = 16*half + 2*g).
// W_ctrl rows: [x(8)|R(32)] = lstm_Wk rows 0..39, h = lstm_Wr rows 40..295,
// rows 296..319 zero pad. W_heads cols: read_W(0..37) | write_W(38..139) | pad.
// ---------------------------------------------------------------------------
__global__ __launch_bounds__(256) void ntm_convert(
    const float* __restrict__ Wk, const float* __restrict__ Wrr,
    const float* __restrict__ rW, const float* __restrict__ wW,
    uint32_t* __restrict__ wsC, uint32_t* __restrict__ wsH) {
  int blk = blockIdx.x, tid = threadIdx.x;
  int lane = tid >> 3, g = tid & 7;
  int half = lane >> 4, ncol = lane & 15;
  if (blk < KC_TILES * NC_TILES) {
    int kt = blk / NC_TILES, nt = blk % NC_TILES;
    int k = kt * 32 + 16 * half + 2 * g;
    int col = nt * 16 + ncol;
    float v0, v1;
    {
      int kk = k;
      v0 = (kk < 40) ? Wk[kk * 1024 + col] : ((kk < 296) ? Wrr[(kk - 40) * 1024 + col] : 0.f);
      kk = k + 1;
      v1 = (kk < 40) ? Wk[kk * 1024 + col] : ((kk < 296) ? Wrr[(kk - 40) * 1024 + col] : 0.f);
    }
    wsC[((size_t)blk * 32 + lane) * 8 + g] = pack_bf16(v0, v1);
  } else {
    int id = blk - KC_TILES * NC_TILES;
    int kt = id / NH_TILES, nt = id % NH_TILES;
    int k = kt * 32 + 16 * half + 2 * g;
    int col = nt * 16 + ncol;
    float v0 = (col < 38) ? rW[k * 38 + col] : ((col < 140) ? wW[k * 102 + (col - 38)] : 0.f);
    float v1 = (col < 38) ? rW[(k + 1) * 38 + col] : ((col < 140) ? wW[(k + 1) * 102 + (col - 38)] : 0.f);
    wsH[((size_t)id * 32 + lane) * 8 + g] = pack_bf16(v0, v1);
  }
}

// ---------------------------------------------------------------------------
// Cooperative NTM addressing (content + interpolate + shift + sharpen).
// All 256 threads participate; b = tid>>4, 16 threads per batch row.
// ---------------------------------------------------------------------------
__device__ __forceinline__ void address_head(
    const float* __restrict__ ho, int colOff,
    float* __restrict__ Wprev, const float* __restrict__ Mblk,
    float* kvec, float* work, float* work2, float* scal, float* red, int tid) {
  {
    int b = tid >> 4, u0 = (tid & 15) * 2;
    kvec[b * 32 + u0]     = tanhf(ho[b * 144 + colOff + u0]);
    kvec[b * 32 + u0 + 1] = tanhf(ho[b * 144 + colOff + u0 + 1]);
  }
  __syncthreads();
  if (tid < B_TILE) {
    int b = tid;
    const float* hb = ho + b * 144 + colOff;
    float beta = softplusf_(hb[32]);
    float gg = sigmoidf_(hb[33]);
    float s0 = hb[34], s1 = hb[35], s2 = hb[36];
    float sm = fmaxf(s0, fmaxf(s1, s2));
    float e0 = expf(s0 - sm), e1 = expf(s1 - sm), e2 = expf(s2 - sm);
    float es = e0 + e1 + e2;
    float gamma = 1.0f + softplusf_(hb[37]);
    float kn = 0.f;
    for (int u = 0; u < 32; ++u) { float kv = kvec[b * 32 + u]; kn += kv * kv; }
    scal[b * 8 + 0] = beta; scal[b * 8 + 1] = gg;
    scal[b * 8 + 2] = e0 / es; scal[b * 8 + 3] = e1 / es; scal[b * 8 + 4] = e2 / es;
    scal[b * 8 + 5] = gamma; scal[b * 8 + 6] = sqrtf(kn);
  }
  __syncthreads();
  int b = tid >> 4, l16 = tid & 15, n0 = l16 * 16;
  float beta = scal[b * 8 + 0], kn = scal[b * 8 + 6];
  float lmax = -3.0e38f;
  for (int nn = 0; nn < 16; ++nn) {
    int n = n0 + nn;
    const float* mrow = Mblk + ((size_t)b * N_SLOTS + n) * M_UNITS;
    float dot = 0.f, nrm = 0.f;
#pragma unroll
    for (int u = 0; u < 32; ++u) { float m = mrow[u]; dot += m * kvec[b * 32 + u]; nrm += m * m; }
    float s = beta * dot / (sqrtf(nrm) * kn + EPSV);
    work[b * 256 + n] = s;
    lmax = fmaxf(lmax, s);
  }
  red[b * 16 + l16] = lmax;
  __syncthreads();
  if (tid < B_TILE) {
    float m = -3.0e38f;
    for (int i = 0; i < 16; ++i) m = fmaxf(m, red[tid * 16 + i]);
    scal[tid * 8 + 7] = m;
  }
  __syncthreads();
  {
    float mx = scal[b * 8 + 7], lsum = 0.f;
    for (int nn = 0; nn < 16; ++nn) {
      int n = n0 + nn;
      float e = expf(work[b * 256 + n] - mx);
      work[b * 256 + n] = e; lsum += e;
    }
    red[b * 16 + l16] = lsum;
  }
  __syncthreads();
  if (tid < B_TILE) {
    float s = 0.f;
    for (int i = 0; i < 16; ++i) s += red[tid * 16 + i];
    scal[tid * 8 + 7] = s;
  }
  __syncthreads();
  {
    float gg = scal[b * 8 + 1];
    float inv = 1.0f / scal[b * 8 + 7];
    for (int nn = 0; nn < 16; ++nn) {
      int n = n0 + nn;
      float wc = work[b * 256 + n] * inv;
      work2[b * 256 + n] = gg * wc + (1.0f - gg) * Wprev[b * 256 + n];
    }
  }
  __syncthreads();
  {
    float s0 = scal[b * 8 + 2], s1 = scal[b * 8 + 3], s2 = scal[b * 8 + 4];
    float gamma = scal[b * 8 + 5];
    float lsum = 0.f;
    for (int nn = 0; nn < 16; ++nn) {
      int n = n0 + nn, np = (n + 1) & 255, nm = (n + 255) & 255;
      // w_s[n] = s0*wg[n+1] + s1*wg[n] + s2*wg[n-1]  (roll(j-1))
      float ws = s0 * work2[b * 256 + np] + s1 * work2[b * 256 + n] + s2 * work2[b * 256 + nm];
      float wp = (ws > 0.f) ? powf(ws, gamma) : 0.f;
      work[b * 256 + n] = wp; lsum += wp;
    }
    red[b * 16 + l16] = lsum;
  }
  __syncthreads();
  if (tid < B_TILE) {
    float s = 0.f;
    for (int i = 0; i < 16; ++i) s += red[tid * 16 + i];
    scal[tid * 8 + 7] = s;
  }
  __syncthreads();
  {
    float inv = 1.0f / (scal[b * 8 + 7] + EPSV);
    for (int nn = 0; nn < 16; ++nn) {
      int n = n0 + nn;
      Wprev[b * 256 + n] = work[b * 256 + n] * inv;
    }
  }
  __syncthreads();
}

// ---------------------------------------------------------------------------
// Main persistent NTM kernel: grid = 32 blocks (16 batch rows each).
// ---------------------------------------------------------------------------
__global__ __launch_bounds__(256) void ntm_main(
    const float* __restrict__ X,
    const float* __restrict__ lstm_b,
    const float* __restrict__ read_b,
    const float* __restrict__ write_b,
    const float* __restrict__ outW,
    const float* __restrict__ outB,
    const float* __restrict__ read_init,
    const float* __restrict__ wr_init,
    const float* __restrict__ ww_init,
    const uint32_t* __restrict__ wsC,
    const uint32_t* __restrict__ wsH,
    float* __restrict__ Mg,
    float* __restrict__ Y) {
  const int tid = threadIdx.x;
  const int wave = tid >> 5, lane = tid & 31;
  const int b0 = blockIdx.x * B_TILE;
  float* Mblk = Mg + (size_t)b0 * N_SLOTS * M_UNITS;

  // 64KB region: z-buffer [16][1024] f32 during controller GEMM/gates,
  // reused as addressing scratch afterwards.
  __shared__ __align__(16) uint8_t s_union[65536];
  __shared__ __align__(16) uint32_t s_avec[16 * 160];  // [x|R|h] bf16 pairs, K=320
  __shared__ __align__(16) uint32_t s_zb16[16 * 128];  // Z (=h) bf16 pairs, K=256
  __shared__ float s_h[16 * 256];
  __shared__ float s_c[16 * 256];
  __shared__ float s_R[16 * 32];
  __shared__ float s_Wr[16 * 256];
  __shared__ float s_Ww[16 * 256];

  float* zbuf  = (float*)s_union;             // [16*1024]
  float* ho    = (float*)s_union;             // [16*144]
  float* kvec  = (float*)(s_union + 9216);    // [16*32]
  float* work  = (float*)(s_union + 11264);   // [16*256]
  float* work2 = (float*)(s_union + 27648);   // [16*256]
  float* scal  = (float*)(s_union + 44032);   // [16*8]
  float* red   = (float*)(s_union + 44544);   // [16*16]
  float* Eb    = (float*)(s_union + 45568);   // [16*32]
  float* Ab    = (float*)(s_union + 47616);   // [16*32]

  // ---- init state ----
  for (int i = tid; i < 16 * 256; i += 256) { s_h[i] = 0.f; s_c[i] = 0.f; }
  {
    int bb = tid >> 4, u0 = (tid & 15) * 2;
    s_R[bb * 32 + u0]     = tanhf(read_init[u0]);
    s_R[bb * 32 + u0 + 1] = tanhf(read_init[u0 + 1]);
  }
  for (int i = tid; i < 16 * 256 * 32; i += 256) Mblk[i] = 1e-6f;
  work[tid] = wr_init[tid];
  work2[tid] = ww_init[tid];
  __syncthreads();
  if (tid == 0) {
    float m1 = -3e38f, m2 = -3e38f;
    for (int i = 0; i < 256; ++i) { m1 = fmaxf(m1, work[i]); m2 = fmaxf(m2, work2[i]); }
    float sa = 0.f, sb = 0.f;
    for (int i = 0; i < 256; ++i) { sa += expf(work[i] - m1); sb += expf(work2[i] - m2); }
    red[0] = m1; red[1] = sa; red[2] = m2; red[3] = sb;
  }
  __syncthreads();
  {
    float e1 = expf(work[tid] - red[0]) / red[1];
    float e2 = expf(work2[tid] - red[2]) / red[3];
    for (int bb = 0; bb < 16; ++bb) { s_Wr[bb * 256 + tid] = e1; s_Ww[bb * 256 + tid] = e2; }
  }
  __syncthreads();

  for (int t = 0; t < T_STEPS; ++t) {
    // P1: pack a_vec = [x_t | R | h] into bf16 (K padded 296->320)
    for (int idx = tid; idx < 16 * 160; idx += 256) {
      int row = idx / 160, w = idx % 160;
      int k0 = w * 2;
      float v[2];
#pragma unroll
      for (int d = 0; d < 2; ++d) {
        int k = k0 + d;
        float x;
        if (k < 8)        x = X[(((size_t)(b0 + row)) * T_STEPS + t) * N_IN + k];
        else if (k < 40)  x = s_R[row * 32 + (k - 8)];
        else if (k < 296) x = s_h[row * 256 + (k - 40)];
        else              x = 0.f;
        v[d] = x;
      }
      s_avec[row * 160 + w] = pack_bf16(v[0], v[1]);
    }
    __syncthreads();

    // P2: controller GEMM z = [16,320] x [320,1024], wave w owns N cols [w*128, w*128+128)
    {
      v8f zero = {0.f, 0.f, 0.f, 0.f, 0.f, 0.f, 0.f, 0.f};
      v8f acc[8];
#pragma unroll
      for (int j = 0; j < 8; ++j) acc[j] = zero;
      for (int kt = 0; kt < KC_TILES; ++kt) {
        Frag a = load_a_lds(s_avec, 160, kt, lane);
        if (kt + 1 < KC_TILES)
          __builtin_prefetch(wsC + ((size_t)((kt + 1) * NC_TILES + wave * 8) * 256) + lane * 8, 0, 0);
#pragma unroll
        for (int j = 0; j < 8; ++j) {
          int nt = wave * 8 + j;
          Frag bf = load_b_glb(wsC + ((size_t)(kt * NC_TILES + nt) * 256), lane);
          acc[j] = __builtin_amdgcn_wmma_f32_16x16x32_bf16(
              false, a.v, false, bf.v, (short)0, acc[j], false, false);
        }
      }
      int half = lane >> 4, nc = lane & 15;
#pragma unroll
      for (int j = 0; j < 8; ++j) {
        int col = (wave * 8 + j) * 16 + nc;
#pragma unroll
        for (int r = 0; r < 8; ++r)
          zbuf[(r + 8 * half) * 1024 + col] = acc[j][r];
      }
    }
    __syncthreads();

    // P3: LSTM gates (keras order i,f,g,o); h in (-1,1) so clip(h,±20)==h
    {
      int row = tid >> 4, u0 = (tid & 15) * 16;
      for (int uu = 0; uu < 16; uu += 2) {
        float hh[2];
#pragma unroll
        for (int d = 0; d < 2; ++d) {
          int u = u0 + uu + d;
          float zi = zbuf[row * 1024 + u]       + lstm_b[u];
          float zf = zbuf[row * 1024 + 256 + u] + lstm_b[256 + u];
          float zg = zbuf[row * 1024 + 512 + u] + lstm_b[512 + u];
          float zo = zbuf[row * 1024 + 768 + u] + lstm_b[768 + u];
          float cc = sigmoidf_(zf) * s_c[row * 256 + u] + sigmoidf_(zi) * tanhf(zg);
          float hv = sigmoidf_(zo) * tanhf(cc);
          s_c[row * 256 + u] = cc;
          s_h[row * 256 + u] = hv;
          hh[d] = hv;
        }
        s_zb16[row * 128 + (u0 + uu) / 2] = pack_bf16(hh[0], hh[1]);
      }
    }
    __syncthreads();

    // P4: head GEMM ho = Z[16,256] x [256,144] (+ biases)
    {
      int half = lane >> 4, nc = lane & 15;
      int nreps = (wave == 0) ? 2 : 1;
      for (int rep = 0; rep < nreps; ++rep) {
        int nt = (rep == 0) ? wave : 8;
        v8f acc = {0.f, 0.f, 0.f, 0.f, 0.f, 0.f, 0.f, 0.f};
        for (int kt = 0; kt < KH_TILES; ++kt) {
          Frag a = load_a_lds(s_zb16, 128, kt, lane);
          Frag bf = load_b_glb(wsH + ((size_t)(kt * NH_TILES + nt) * 256), lane);
          acc = __builtin_amdgcn_wmma_f32_16x16x32_bf16(
              false, a.v, false, bf.v, (short)0, acc, false, false);
        }
        int col = nt * 16 + nc;
        float bias = (col < 38) ? read_b[col] : ((col < 140) ? write_b[col - 38] : 0.f);
#pragma unroll
        for (int r = 0; r < 8; ++r)
          ho[(r + 8 * half) * 144 + col] = acc[r] + bias;
      }
    }
    __syncthreads();

    // P5: read head addressing (pre-write M) + read vector R
    address_head(ho, 0, s_Wr, Mblk, kvec, work, work2, scal, red, tid);
    {
      int bb = tid >> 4, u0 = (tid & 15) * 2;
#pragma unroll
      for (int d = 0; d < 2; ++d) {
        int u = u0 + d;
        float a = 0.f;
        for (int n = 0; n < 256; ++n)
          a += s_Wr[bb * 256 + n] * Mblk[((size_t)bb * N_SLOTS + n) * M_UNITS + u];
        s_R[bb * 32 + u] = a;
      }
    }
    __syncthreads();

    // P6: write head addressing (pre-write M)
    address_head(ho, ADDRW, s_Ww, Mblk, kvec, work, work2, scal, red, tid);

    // P7: erase/add memory update
    {
      int bb = tid >> 4, u0 = (tid & 15) * 2;
#pragma unroll
      for (int d = 0; d < 2; ++d) {
        int u = u0 + d;
        Eb[bb * 32 + u] = sigmoidf_(ho[bb * 144 + 76 + u]);
        Ab[bb * 32 + u] = tanhf(ho[bb * 144 + 108 + u]);
      }
    }
    __syncthreads();
    {
      int bb = tid >> 4, n0 = (tid & 15) * 16;
      for (int nn = 0; nn < 16; ++nn) {
        int n = n0 + nn;
        float wn = s_Ww[bb * 256 + n];
        float* mrow = Mblk + ((size_t)bb * N_SLOTS + n) * M_UNITS;
#pragma unroll
        for (int u = 0; u < 32; ++u) {
          float m = mrow[u];
          mrow[u] = m * (1.0f - wn * Eb[bb * 32 + u]) + wn * Ab[bb * 32 + u];
        }
      }
    }
    __syncthreads();

    // P8: output Y = clip([R|Z] @ outW + outB)
    if (tid < 128) {
      int bb = tid >> 3, o = tid & 7;
      float acc = outB[o];
      for (int u = 0; u < 32; ++u)  acc += s_R[bb * 32 + u] * outW[u * 8 + o];
      for (int u = 0; u < 256; ++u) acc += s_h[bb * 256 + u] * outW[(32 + u) * 8 + o];
      acc = fminf(fmaxf(acc, -CLIPV), CLIPV);
      Y[(((size_t)(b0 + bb)) * T_STEPS + t) * N_OUT + o] = acc;
    }
    __syncthreads();
  }
}

extern "C" void kernel_launch(void* const* d_in, const int* in_sizes, int n_in,
                              void* d_out, int out_size, void* d_ws, size_t ws_size,
                              hipStream_t stream) {
  const float* X         = (const float*)d_in[0];
  const float* lstm_Wk   = (const float*)d_in[1];
  const float* lstm_Wr   = (const float*)d_in[2];
  const float* lstm_b    = (const float*)d_in[3];
  const float* read_W    = (const float*)d_in[4];
  const float* read_b    = (const float*)d_in[5];
  const float* write_W   = (const float*)d_in[6];
  const float* write_b   = (const float*)d_in[7];
  const float* out_W     = (const float*)d_in[8];
  const float* out_b     = (const float*)d_in[9];
  const float* read_init = (const float*)d_in[10];
  const float* wr_init   = (const float*)d_in[11];
  const float* ww_init   = (const float*)d_in[12];

  // ws layout: [0, 640K) ctrl frags | [640K, 712K) head frags | [1M, 17M) M
  uint32_t* wsC = (uint32_t*)d_ws;
  uint32_t* wsH = (uint32_t*)((char*)d_ws + (size_t)KC_TILES * NC_TILES * 1024);
  float* Mg = (float*)((char*)d_ws + (1u << 20));

  ntm_convert<<<KC_TILES * NC_TILES + KH_TILES * NH_TILES, 256, 0, stream>>>(
      lstm_Wk, lstm_Wr, read_W, write_W, wsC, wsH);
  ntm_main<<<512 / B_TILE, 256, 0, stream>>>(
      X, lstm_b, read_b, write_b, out_W, out_b, read_init, wr_init, ww_init,
      wsC, wsH, Mg, (float*)d_out);
}
